// AE_layer_22686017257949
// MI455X (gfx1250) — matrix operand order repaired and checked by reading, not compiled, not run on hardware
//
#include <hip/hip_runtime.h>
#include <hip/hip_bf16.h>

#define N_NODES   50000
#define E_EDGES   800000
#define E_TOT     (E_EDGES + N_NODES)   // 850000 (with self loops)
#define IN_F      256
#define HEADS     4
#define OUT_F     64
#define HF        (HEADS * OUT_F)       // 256
#define WCOLS     (2 * HF)              // 512: [Wl | Wr]
#define NEG_SLOPE 0.2f
#define EPS_GN    1e-5f

typedef __attribute__((ext_vector_type(16))) __bf16          v16bf;
typedef __attribute__((ext_vector_type(16))) unsigned short  v16us;
typedef __attribute__((ext_vector_type(8)))  unsigned int    v8u;
typedef __attribute__((ext_vector_type(8)))  float           v8f;

__device__ __forceinline__ unsigned short f2bf(float f) {
    unsigned int u = __builtin_bit_cast(unsigned int, f);
    unsigned int r = u + 0x7FFFu + ((u >> 16) & 1u);   // round-to-nearest-even
    return (unsigned short)(r >> 16);
}
// order-preserving float -> uint encoding for atomic max
__device__ __forceinline__ unsigned int encf(float f) {
    unsigned int u = __builtin_bit_cast(unsigned int, f);
    return (u & 0x80000000u) ? ~u : (u | 0x80000000u);
}
__device__ __forceinline__ float decf(unsigned int u) {
    u = (u & 0x80000000u) ? (u & 0x7FFFFFFFu) : ~u;
    return __builtin_bit_cast(float, u);
}

// ---------- conversions ----------
__global__ void cvt_x_bf16(const float* __restrict__ X, unsigned short* __restrict__ Xb) {
    long long i = (long long)blockIdx.x * 256 + threadIdx.x;
    if (i < (long long)N_NODES * IN_F) Xb[i] = f2bf(X[i]);
}

// Pack [Wl|Wr] (256x512) directly into WMMA B-fragment lane order:
// Wp[kc][nt][lane][j]  (kc: K/32 chunk, nt: 16-col tile, lane 0..31, j 0..15)
// element = W[ kc*32 + (lane>>4)*16 + j ][ nt*16 + (lane&15) ]
__global__ void cvt_w_pack(const float* __restrict__ Wl, const float* __restrict__ Wr,
                           unsigned short* __restrict__ Wp) {
    int i = blockIdx.x * 256 + threadIdx.x;           // 8*32*32*16 = 131072
    if (i >= (IN_F / 32) * 32 * 32 * 16) return;
    int j    = i & 15;
    int lane = (i >> 4) & 31;
    int nt   = (i >> 9) & 31;
    int kc   = i >> 14;
    int K    = kc * 32 + (lane >> 4) * 16 + j;
    int Nc   = nt * 16 + (lane & 15);
    float v  = (Nc < HF) ? Wl[K * HF + Nc] : Wr[K * HF + (Nc - HF)];
    Wp[i] = f2bf(v);
}

// ---------- WMMA GEMM: Y[N,512] = X[N,256] @ [Wl|Wr][256,512] ----------
// block = 256 threads = 8 waves; block owns a 16-row M tile, each wave 64 cols.
// K loop fully unrolled: 32 static v_wmma per block.
__global__ void gemm_wmma(const unsigned short* __restrict__ Xb,
                          const unsigned short* __restrict__ Wp,
                          float* __restrict__ xl, float* __restrict__ xr) {
    __shared__ unsigned int lAu[16 * 16];             // 16 rows x 32 bf16
    const int tid  = threadIdx.x;
    const int lane = tid & 31;
    const int wave = tid >> 5;
    const int m0   = blockIdx.x * 16;
    const int row  = lane & 15;
    const int half = lane >> 4;
    const int col  = lane & 15;
    const int ar   = tid >> 4;                        // A-load row 0..15
    const int ap   = tid & 15;                        // A-load dword pair

    v8f acc[4] = {};

#pragma unroll
    for (int kc = 0; kc < IN_F / 32; ++kc) {
        const int k0 = kc * 32;
        __syncthreads();
        // cooperative A tile load: one dword (2 bf16) per thread
        lAu[tid] = *(const unsigned int*)(Xb + (size_t)(m0 + ar) * IN_F + k0 + ap * 2);
        if (kc + 1 < IN_F / 32)                       // prefetch next A tile
            __builtin_prefetch(Xb + (size_t)(m0 + ar) * IN_F + k0 + 32 + ap * 2, 0, 3);
        __syncthreads();

        // A fragment in contiguous K order K = half*16 + s, as packed in Wp (two b128 LDS loads)
        const unsigned short* lA = (const unsigned short*)lAu;   // [16][32]
        v16us au;
#pragma unroll
        for (int j = 0; j < 8; ++j) {
            au[j]     = lA[row * 32 + half * 16 + j];       // K = half*16 + j
            au[j + 8] = lA[row * 32 + half * 16 + 8 + j];   // K = half*16 + 8 + j
        }
        v16bf a = __builtin_bit_cast(v16bf, au);

#pragma unroll
        for (int t = 0; t < 4; ++t) {
            const int nt = wave * 4 + t;              // 16-col tile index (0..31)
            // pre-packed B fragment: one contiguous 32B load per lane
            v8u bu = *(const v8u*)(Wp + (((size_t)kc * 32 + nt) * 32 + lane) * 16);
            v16bf b = __builtin_bit_cast(v16bf, bu);
            acc[t] = __builtin_amdgcn_wmma_f32_16x16x32_bf16(
                false, a, false, b, (short)0, acc[t], false, false);
        }
    }

#pragma unroll
    for (int t = 0; t < 4; ++t) {
        int nt = wave * 64 + t * 16;
        float* dst = (nt < HF) ? xl : xr;
        int nbase  = (nt < HF) ? nt : nt - HF;
#pragma unroll
        for (int r = 0; r < 8; ++r) {
            int m = m0 + half * 8 + r;                // C/D layout
            dst[(size_t)m * HF + nbase + col] = acc[t][r];
        }
    }
}

// ---------- zero fill ----------
__global__ void fill0_u32(unsigned int* __restrict__ p, long long count) {
    long long i = (long long)blockIdx.x * 256 + threadIdx.x;
    if (i < count) p[i] = 0u;
}

// ---------- edge pass 1: logits + segment max (one wave per edge) ----------
__global__ void edge_logits(const int* __restrict__ E0, const int* __restrict__ E1,
                            const float* __restrict__ xl, const float* __restrict__ xr,
                            const float* __restrict__ att,
                            float* __restrict__ logits, unsigned int* __restrict__ menc) {
    __shared__ float s_att[HF];
    s_att[threadIdx.x] = att[threadIdx.x];
    __syncthreads();
    int wave = threadIdx.x >> 5, lane = threadIdx.x & 31;
    long long e = (long long)blockIdx.x * 8 + wave;
    if (e >= E_TOT) return;
    int s, d;
    if (e < E_EDGES) { s = E0[e]; d = E1[e]; } else { s = d = (int)(e - E_EDGES); }
    int h = lane >> 3, k = lane & 7;
    int base = h * OUT_F + k * 8;
    const float4* pl = (const float4*)(xl + (size_t)s * HF + base);
    const float4* pr = (const float4*)(xr + (size_t)d * HF + base);
    float acc = 0.f;
#pragma unroll
    for (int q = 0; q < 2; ++q) {
        float4 a4 = pl[q], b4 = pr[q];
        float v[4] = {a4.x + b4.x, a4.y + b4.y, a4.z + b4.z, a4.w + b4.w};
#pragma unroll
        for (int j = 0; j < 4; ++j) {
            float t = v[j] > 0.f ? v[j] : NEG_SLOPE * v[j];
            acc += t * s_att[base + q * 4 + j];
        }
    }
    acc += __shfl_xor(acc, 1, 32);
    acc += __shfl_xor(acc, 2, 32);
    acc += __shfl_xor(acc, 4, 32);
    if (k == 0) {
        logits[e * HEADS + h] = acc;
        atomicMax(menc + (size_t)d * HEADS + h, encf(acc));
    }
}

// ---------- edge pass 2a: p = exp(logit - m[dst]), denom += p ----------
__global__ void edge_exp(const int* __restrict__ E1, float* __restrict__ lp,
                         const unsigned int* __restrict__ menc, float* __restrict__ denom) {
    long long i = (long long)blockIdx.x * 256 + threadIdx.x;
    if (i >= (long long)E_TOT * HEADS) return;
    long long e = i >> 2;
    int h = (int)(i & 3);
    int d = (e < E_EDGES) ? E1[e] : (int)(e - E_EDGES);
    float m = decf(menc[(size_t)d * HEADS + h]);
    float p = __expf(lp[i] - m);
    lp[i] = p;
    atomicAdd(denom + (size_t)d * HEADS + h, p);
}

// ---------- edge pass 2b: acc[dst] += p * xl[src]  (block per edge, 256 lanes = H*C) --
__global__ void edge_accum(const int* __restrict__ E0, const int* __restrict__ E1,
                           const float* __restrict__ xl, const float* __restrict__ pbuf,
                           float* __restrict__ acc) {
    long long e = blockIdx.x;
    int tid = threadIdx.x;                         // tid = h*64 + c
    int s, d;
    if (e < E_EDGES) { s = E0[e]; d = E1[e]; } else { s = d = (int)(e - E_EDGES); }
    float p = pbuf[e * HEADS + (tid >> 6)];
    atomicAdd(acc + (size_t)d * HF + tid, p * xl[(size_t)s * HF + tid]);
}

// ---------- node pass: mean over heads + bias, partial column sums for mu ----------
__global__ void node_combine(const float* __restrict__ acc, const float* __restrict__ denom,
                             const float* __restrict__ bias, float* __restrict__ o,
                             float* __restrict__ musum) {
    __shared__ float red[256];
    int tid = threadIdx.x;
    int n = blockIdx.x * 4 + (tid >> 6);
    int c = tid & 63;
    float v = 0.f;
#pragma unroll
    for (int h = 0; h < HEADS; ++h)
        v += acc[(size_t)n * HF + h * OUT_F + c] / (denom[(size_t)n * HEADS + h] + 1e-16f);
    v = v * (1.0f / HEADS) + bias[c];
    o[(size_t)n * OUT_F + c] = v;
    red[tid] = v;
    __syncthreads();
    if (tid < 128) red[tid] += red[tid + 128];
    __syncthreads();
    if (tid < 64) atomicAdd(musum + tid, red[tid] + red[tid + 64]);
}

// ---------- center with mean_scale, partial column sums of squares ----------
__global__ void node_center(float* __restrict__ o, const float* __restrict__ musum,
                            const float* __restrict__ alpha, float* __restrict__ varsum) {
    __shared__ float red[256];
    int tid = threadIdx.x;
    int n = blockIdx.x * 4 + (tid >> 6);
    int c = tid & 63;
    float mu = musum[c] * (1.0f / N_NODES);
    float xc = o[(size_t)n * OUT_F + c] - alpha[c] * mu;
    o[(size_t)n * OUT_F + c] = xc;
    red[tid] = xc * xc;
    __syncthreads();
    if (tid < 128) red[tid] += red[tid + 128];
    __syncthreads();
    if (tid < 64) atomicAdd(varsum + tid, red[tid] + red[tid + 64]);
}

// ---------- final normalize ----------
__global__ void node_final(const float* __restrict__ o, const float* __restrict__ varsum,
                           const float* __restrict__ gamma, const float* __restrict__ beta,
                           float* __restrict__ y) {
    long long i = (long long)blockIdx.x * 256 + threadIdx.x;
    if (i >= (long long)N_NODES * OUT_F) return;
    int c = (int)(i & 63);
    float var = varsum[c] * (1.0f / N_NODES);
    y[i] = gamma[c] * o[i] * rsqrtf(var + EPS_GN) + beta[c];
}

extern "C" void kernel_launch(void* const* d_in, const int* in_sizes, int n_in,
                              void* d_out, int out_size, void* d_ws, size_t ws_size,
                              hipStream_t stream) {
    (void)in_sizes; (void)n_in; (void)out_size; (void)ws_size;
    const float* X    = (const float*)d_in[0];
    const int*   E    = (const int*)d_in[1];          // [2, E_EDGES]
    const float* Wl   = (const float*)d_in[2];
    const float* Wr   = (const float*)d_in[3];
    const float* att  = (const float*)d_in[4];
    const float* bias = (const float*)d_in[5];
    const float* gnw  = (const float*)d_in[6];
    const float* gnb  = (const float*)d_in[7];
    const float* gnms = (const float*)d_in[8];
    float* y = (float*)d_out;
    const int* E0 = E;
    const int* E1 = E + E_EDGES;

    // workspace carve-out (all 256-byte aligned)
    char* ws = (char*)d_ws;
    size_t off = 0;
    auto carve = [&](size_t bytes) { char* p = ws + off; off += (bytes + 255) & ~(size_t)255; return p; };
    unsigned short* Xb     = (unsigned short*)carve((size_t)N_NODES * IN_F * 2);   // 25.6 MB
    unsigned short* Wp     = (unsigned short*)carve((size_t)IN_F * WCOLS * 2);     // 0.26 MB packed
    float*          xl     = (float*)carve((size_t)N_NODES * HF * 4);              // 51.2 MB
    float*          xr     = (float*)carve((size_t)N_NODES * HF * 4);              // 51.2 MB (reused as acc)
    float*          logits = (float*)carve((size_t)E_TOT * HEADS * 4);             // 13.6 MB
    unsigned int*   menc   = (unsigned int*)carve((size_t)N_NODES * HEADS * 4);    // zero-region start
    float*          denom  = (float*)carve((size_t)N_NODES * HEADS * 4);
    float*          musum  = (float*)carve(64 * 4);
    float*          varsum = (float*)carve(64 * 4);
    float*          o      = (float*)carve((size_t)N_NODES * OUT_F * 4);           // 12.8 MB

    // 0) zero m/denom/musum/varsum (contiguous: sizes are multiples of 256B)
    long long zcnt = (long long)N_NODES * HEADS * 2 + 128;
    fill0_u32<<<(unsigned)((zcnt + 255) / 256), 256, 0, stream>>>(menc, zcnt);

    // 1) convert / pack to bf16
    cvt_x_bf16<<<(unsigned)(((long long)N_NODES * IN_F + 255) / 256), 256, 0, stream>>>(X, Xb);
    cvt_w_pack<<<(IN_F * WCOLS + 255) / 256, 256, 0, stream>>>(Wl, Wr, Wp);

    // 2) WMMA GEMM -> xl, xr
    gemm_wmma<<<N_NODES / 16, 256, 0, stream>>>(Xb, Wp, xl, xr);

    // 3) edge logits + segment max
    edge_logits<<<(E_TOT + 7) / 8, 256, 0, stream>>>(E0, E1, xl, xr, att, logits, menc);

    // 4) zero accumulation buffer (reuse xr; pass 3 no longer needs it)
    long long acnt = (long long)N_NODES * HF;
    fill0_u32<<<(unsigned)((acnt + 255) / 256), 256, 0, stream>>>((unsigned int*)xr, acnt);

    // 5) exp + denom
    long long ecnt = (long long)E_TOT * HEADS;
    edge_exp<<<(unsigned)((ecnt + 255) / 256), 256, 0, stream>>>(E1, logits, menc, denom);

    // 6) weighted scatter-add into acc (= xr buffer)
    edge_accum<<<E_TOT, 256, 0, stream>>>(E0, E1, xl, logits, xr);

    // 7) head-mean + bias, mu partials
    node_combine<<<N_NODES / 4, 256, 0, stream>>>(xr, denom, bias, o, musum);

    // 8) center, variance partials
    node_center<<<N_NODES / 4, 256, 0, stream>>>(o, musum, gnms, varsum);

    // 9) normalize
    node_final<<<(unsigned)(((long long)N_NODES * OUT_F + 255) / 256), 256, 0, stream>>>(
        o, varsum, gnw, gnb, y);
}
